// GroupAttention_7885559955565
// MI455X (gfx1250) — matrix operand order repaired and checked by reading, not compiled, not run on hardware
//
#include <hip/hip_runtime.h>

// ---------------------------------------------------------------------------
// Types / helpers
// ---------------------------------------------------------------------------
typedef __attribute__((ext_vector_type(16))) __bf16 v16bf;
typedef __attribute__((ext_vector_type(8)))  float  v8f;

union Frag16 { v16bf v; uint4 u[2]; };
union Pk8    { uint4 u; __bf16 e[8]; };

// Native f32 -> bf16 (compiler emits v_cvt_pk_bf16_f32)
__device__ __forceinline__ __bf16 f2bf(float f) { return (__bf16)f; }

__device__ __forceinline__ v8f zero8() {
  v8f z;
#pragma unroll
  for (int i = 0; i < 8; ++i) z[i] = 0.0f;
  return z;
}

__device__ __forceinline__ v8f wmma_bf16(v16bf a, v16bf b, v8f c) {
  // D(f32 16x16) = A(bf16 16x32) * B(bf16 32x16) + C
  return __builtin_amdgcn_wmma_f32_16x16x32_bf16(false, a, false, b,
                                                 (short)0, c, false, false);
}

// LDS byte offset of a generic shared pointer (addr[31:0] is the LDS offset)
__device__ __forceinline__ unsigned lds_off(const void* p) {
  return (unsigned)(unsigned long long)p;
}

// CDNA5 async LDS DMA: copies 16 bytes global -> LDS per lane (ASYNCcnt)
__device__ __forceinline__ void async_load_b128(unsigned lds, const void* g) {
  asm volatile("global_load_async_to_lds_b128 %0, %1, off"
               :: "v"(lds), "v"(g)
               : "memory");
}

__device__ __forceinline__ void wait_async0() {
  asm volatile("s_wait_asynccnt 0" ::: "memory");
}

// Problem constants (fixed by the reference)
#define C_DIM   512
#define NT3     1536           // 3*C
#define MTOK    100352         // B*N = 32*3136
#define SPAD    64             // window tokens padded 49 -> 64
#define LDSTRIDE 40            // bf16 elems per LDS tile row (16B aligned, bank-safe)

// ---------------------------------------------------------------------------
// Kernel 0a: weights f32 -> bf16
// ---------------------------------------------------------------------------
__global__ __launch_bounds__(256) void cvt_weights_kernel(
    const float* __restrict__ wqkv, const float* __restrict__ wproj,
    __bf16* __restrict__ oq, __bf16* __restrict__ op) {
  int i = blockIdx.x * 256 + threadIdx.x;
  if (i < 786432) oq[i] = f2bf(wqkv[i]);
  int j = i - 786432;
  if (j >= 0 && j < 262144) op[j] = f2bf(wproj[j]);
}

// ---------------------------------------------------------------------------
// Kernel 0b: x f32 -> bf16 (once; QKV GEMM then reads half the bytes, 12x)
// ---------------------------------------------------------------------------
__global__ __launch_bounds__(256) void cvt_x_kernel(
    const float* __restrict__ x, __bf16* __restrict__ xb) {
  const size_t i = ((size_t)blockIdx.x * 256 + threadIdx.x) * 8;
  const float4* s = (const float4*)(x + i);
  const float4 f0 = s[0], f1 = s[1];
  Pk8 o;
  o.e[0] = f2bf(f0.x); o.e[1] = f2bf(f0.y);
  o.e[2] = f2bf(f0.z); o.e[3] = f2bf(f0.w);
  o.e[4] = f2bf(f1.x); o.e[5] = f2bf(f1.y);
  o.e[6] = f2bf(f1.z); o.e[7] = f2bf(f1.w);
  *(uint4*)(xb + i) = o.u;
}

// ---------------------------------------------------------------------------
// Shared GEMM mainloop: 128x128 C-tile per block, 8 waves (4Mx2N),
// each wave 2x4 WMMA tiles, K stepped by 32.
// Double-buffered LDS, tiles staged with GLOBAL_LOAD_ASYNC_TO_LDS_B128 so
// the DMA for tile k+32 overlaps the 8 WMMAs on tile k.
// A is MxK row-major bf16; B is NxK row-major (weights) bf16.
// ---------------------------------------------------------------------------
__device__ __forceinline__ void gemm_tile(const __bf16* __restrict__ Abf,
                                          const __bf16* __restrict__ Bw,
                                          int K, int rowBase, int colBase,
                                          __bf16* sA0, __bf16* sB0,
                                          __bf16* sA1, __bf16* sB1,
                                          v8f acc[2][4]) {
  const int tid  = threadIdx.x;
  const int lane = tid & 31;
  const int wave = tid >> 5;
  const int wm = wave >> 1, wn = wave & 1;
  const int half = lane >> 4, l15 = lane & 15;
  const int srow = tid >> 1, sseg = tid & 1;     // staging: 2 threads per row

  const __bf16* aSrc = Abf + (size_t)(rowBase + srow) * K + sseg * 16;
  const __bf16* bSrc = Bw  + (size_t)(colBase + srow) * K + sseg * 16;
  const int soff = srow * LDSTRIDE + sseg * 16;
  const unsigned aD[2] = { lds_off(&sA0[soff]), lds_off(&sA1[soff]) };
  const unsigned bD[2] = { lds_off(&sB0[soff]), lds_off(&sB1[soff]) };
  const __bf16* sAbuf[2] = { sA0, sA1 };
  const __bf16* sBbuf[2] = { sB0, sB1 };

  // ---- prologue: DMA first tile into buffer 0 ----
  async_load_b128(aD[0],      aSrc);
  async_load_b128(aD[0] + 16, aSrc + 8);
  async_load_b128(bD[0],      bSrc);
  async_load_b128(bD[0] + 16, bSrc + 8);
  __builtin_prefetch(aSrc + 32, 0, 1);
  __builtin_prefetch(bSrc + 32, 0, 1);

  int cur = 0;
  for (int k0 = 0; k0 < K; k0 += 32, cur ^= 1) {
    wait_async0();                 // copies for buffer `cur` have landed
    __syncthreads();               // ... in every wave; prev reads of cur^1 done

    if (k0 + 32 < K) {             // fire DMA for the next tile (other buffer)
      const int nxt = cur ^ 1;
      async_load_b128(aD[nxt],      aSrc + k0 + 32);
      async_load_b128(aD[nxt] + 16, aSrc + k0 + 40);
      async_load_b128(bD[nxt],      bSrc + k0 + 32);
      async_load_b128(bD[nxt] + 16, bSrc + k0 + 40);
      if (k0 + 64 < K) {           // keep L2 one tile ahead of the DMA
        __builtin_prefetch(aSrc + k0 + 64, 0, 1);
        __builtin_prefetch(bSrc + k0 + 64, 0, 1);
      }
    }

    // ---- fragments + WMMA on the current buffer ----
    const __bf16* cA = sAbuf[cur];
    const __bf16* cB = sBbuf[cur];
    Frag16 aF[2], bF[4];
#pragma unroll
    for (int mt = 0; mt < 2; ++mt) {
      // A 16x32: lane half -> contiguous K {h*16 .. h*16+15}
      const uint4* p =
          (const uint4*)&cA[(wm * 32 + mt * 16 + l15) * LDSTRIDE + half * 16];
      aF[mt].u[0] = p[0]; aF[mt].u[1] = p[1];
    }
#pragma unroll
    for (int nt = 0; nt < 4; ++nt) {
      // B 32x16: lane half -> contiguous K {h*16 .. h*16+15}
      const uint4* p =
          (const uint4*)&cB[(wn * 64 + nt * 16 + l15) * LDSTRIDE + half * 16];
      bF[nt].u[0] = p[0]; bF[nt].u[1] = p[1];
    }
#pragma unroll
    for (int mt = 0; mt < 2; ++mt)
#pragma unroll
      for (int nt = 0; nt < 4; ++nt)
        acc[mt][nt] = wmma_bf16(aF[mt].v, bF[nt].v, acc[mt][nt]);
  }
}

// ---------------------------------------------------------------------------
// Kernel 1: QKV GEMM + 2D RoPE epilogue, scatter to padded (B,G,h,64,64) bf16
// ---------------------------------------------------------------------------
__global__ __launch_bounds__(256) void qkv_rope_kernel(
    const __bf16* __restrict__ xb, const __bf16* __restrict__ wqkv,
    __bf16* __restrict__ qo, __bf16* __restrict__ ko, __bf16* __restrict__ vo) {
  __shared__ __bf16 sA[2][128 * LDSTRIDE];
  __shared__ __bf16 sB[2][128 * LDSTRIDE];
  v8f acc[2][4];
#pragma unroll
  for (int mt = 0; mt < 2; ++mt)
#pragma unroll
    for (int nt = 0; nt < 4; ++nt) acc[mt][nt] = zero8();

  const int rowBase = blockIdx.x * 128;
  const int colBase = blockIdx.y * 128;
  gemm_tile(xb, wqkv, C_DIM, rowBase, colBase,
            sA[0], sB[0], sA[1], sB[1], acc);

  const int tid = threadIdx.x, lane = tid & 31, wave = tid >> 5;
  const int wm = wave >> 1, wn = wave & 1;
  const int half = lane >> 4, l15 = lane & 15;
  const int mB = rowBase + wm * 32;
  const int nB = colBase + wn * 64;
#pragma unroll
  for (int mt = 0; mt < 2; ++mt) {
#pragma unroll
    for (int r = 0; r < 8; ++r) {
      const int m  = mB + mt * 16 + half * 8 + r;     // token id (D-layout row)
      const int b  = m / 3136;
      const int n  = m % 3136;
      const int hh = n / 56, ww = n % 56;
      const int sh = hh % 7, sw = ww % 7;
      const int g  = (hh / 7) * 8 + (ww / 7);
      const int s  = sh * 7 + sw;
      const size_t hb = (size_t)(b * 64 + g) * 8;
#pragma unroll
      for (int nt = 0; nt < 4; ++nt) {
        float val     = acc[mt][nt][r];
        float partner = __shfl_xor(val, 1, 32);       // RoPE pair lives in lane^1
        const int d     = nB + nt * 16 + l15;         // 0..1535
        const int which = d >> 9;                     // 0=q 1=k 2=v
        const int rem   = d & 511;
        const int hIdx  = rem >> 6;
        const int e     = rem & 63;
        float outv = val;
        if (which < 2) {
          const int   j    = e >> 2;                  // freq index 0..15
          const int   axis = (e >> 1) & 1;            // 0: x-pos, 1: y-pos
          const float pos  = (float)(axis ? sh : sw);
          const float freq = __expf(-0.57564627f * (float)j);  // 10000^(-j/16)
          const float ang  = pos * freq;
          const float c = __cosf(ang), sn = __sinf(ang);
          outv = ((e & 1) == 0) ? (val * c - partner * sn)
                                : (val * c + partner * sn);
        }
        __bf16* dst = (which == 0) ? qo : (which == 1) ? ko : vo;
        dst[((hb + hIdx) * SPAD + s) * 64 + e] = f2bf(outv);
      }
    }
  }
}

// ---------------------------------------------------------------------------
// Kernel 2: windowed attention, one block per (b, g, head), 4 waves.
// scores(64x64, cols>=49 masked) -> softmax -> P@V, all bf16 WMMA.
// ---------------------------------------------------------------------------
__global__ __launch_bounds__(128) void attn_kernel(
    const __bf16* __restrict__ q, const __bf16* __restrict__ k,
    const __bf16* __restrict__ v, __bf16* __restrict__ ao) {
  __shared__ __bf16 sP [64 * 88];   // softmax probs, A-fragment friendly
  __shared__ __bf16 sVT[64 * 88];   // V transposed: sVT[d][t]

  const int bid = blockIdx.x;
  const int h = bid & 7, g = (bid >> 3) & 63, b = bid >> 9;
  const size_t base = ((size_t)(b * 64 + g) * 8 + h) * (SPAD * 64);
  const __bf16* qb = q + base;
  const __bf16* kb = k + base;
  const __bf16* vb = v + base;

  const int tid = threadIdx.x, lane = tid & 31, wave = tid >> 5;
  const int half = lane >> 4, l15 = lane & 15;

  // ---- stage V^T into LDS (coalesced b128 reads, b16 scatter) ----
  for (int i = tid; i < 512; i += 128) {
    const int t = i >> 3, cg = i & 7;
    Pk8 pk; pk.u = *(const uint4*)(vb + t * 64 + cg * 8);
#pragma unroll
    for (int j = 0; j < 8; ++j) sVT[(cg * 8 + j) * 88 + t] = pk.e[j];
  }

  // ---- S = q @ k^T (rows: this wave's 16 tokens; cols: all 64 padded) ----
  v8f accS[4];
#pragma unroll
  for (int nt = 0; nt < 4; ++nt) accS[nt] = zero8();
#pragma unroll
  for (int ks = 0; ks < 2; ++ks) {
    Frag16 aF;
    const uint4* ap = (const uint4*)(qb + (wave * 16 + l15) * 64 + ks * 32 + half * 16);
    aF.u[0] = ap[0]; aF.u[1] = ap[1];
#pragma unroll
    for (int nt = 0; nt < 4; ++nt) {
      Frag16 bF;  // B column n = k row (nt*16 + n), contiguous K per lane half
      const uint4* bp = (const uint4*)(kb + (nt * 16 + l15) * 64 + ks * 32 + half * 16);
      bF.u[0] = bp[0]; bF.u[1] = bp[1];
      accS[nt] = wmma_bf16(aF.v, bF.v, accS[nt]);
    }
  }

  // ---- masked row softmax (row spread over 16 lanes of one half) ----
  const int prow = wave * 16 + half * 8;
#pragma unroll
  for (int r = 0; r < 8; ++r) {
    float mx = -1e30f;
#pragma unroll
    for (int nt = 0; nt < 4; ++nt) {
      const int col = nt * 16 + l15;
      const float vv = (col < 49) ? accS[nt][r] * 0.125f : -1e30f;  // SCALE=1/8
      accS[nt][r] = vv;
      mx = fmaxf(mx, vv);
    }
#pragma unroll
    for (int o = 1; o < 16; o <<= 1) mx = fmaxf(mx, __shfl_xor(mx, o, 32));
    float sum = 0.0f;
#pragma unroll
    for (int nt = 0; nt < 4; ++nt) {
      const float e = __expf(accS[nt][r] - mx);
      accS[nt][r] = e; sum += e;
    }
#pragma unroll
    for (int o = 1; o < 16; o <<= 1) sum += __shfl_xor(sum, o, 32);
    const float inv = 1.0f / sum;
#pragma unroll
    for (int nt = 0; nt < 4; ++nt)
      sP[(prow + r) * 88 + nt * 16 + l15] = f2bf(accS[nt][r] * inv);
  }
  __syncthreads();

  // ---- O = P @ V ----
  v8f accO[4];
#pragma unroll
  for (int nt = 0; nt < 4; ++nt) accO[nt] = zero8();
#pragma unroll
  for (int ks = 0; ks < 2; ++ks) {
    Frag16 aF;
    const uint4* ap = (const uint4*)&sP[(wave * 16 + l15) * 88 + ks * 32 + half * 16];
    aF.u[0] = ap[0]; aF.u[1] = ap[1];
#pragma unroll
    for (int nt = 0; nt < 4; ++nt) {
      Frag16 bF;  // B column n = hd dim (nt*16 + n), K = token index
      const uint4* bp = (const uint4*)&sVT[(nt * 16 + l15) * 88 + ks * 32 + half * 16];
      bF.u[0] = bp[0]; bF.u[1] = bp[1];
      accO[nt] = wmma_bf16(aF.v, bF.v, accO[nt]);
    }
  }

  // ---- un-window and store bf16 to (B,N,C) ----
#pragma unroll
  for (int r = 0; r < 8; ++r) {
    const int s = wave * 16 + half * 8 + r;
    if (s < 49) {
      const int hh = (g >> 3) * 7 + s / 7;
      const int ww = (g & 7) * 7 + s % 7;
      const int n  = hh * 56 + ww;
      const size_t ob = ((size_t)b * 3136 + n) * C_DIM + h * 64;
#pragma unroll
      for (int nt = 0; nt < 4; ++nt)
        ao[ob + nt * 16 + l15] = f2bf(accO[nt][r]);
    }
  }
}

// ---------------------------------------------------------------------------
// Kernel 3: output projection GEMM + bias, f32 result
// ---------------------------------------------------------------------------
__global__ __launch_bounds__(256) void proj_kernel(
    const __bf16* __restrict__ ao, const __bf16* __restrict__ wp,
    const float* __restrict__ bias, float* __restrict__ out) {
  __shared__ __bf16 sA[2][128 * LDSTRIDE];
  __shared__ __bf16 sB[2][128 * LDSTRIDE];
  v8f acc[2][4];
#pragma unroll
  for (int mt = 0; mt < 2; ++mt)
#pragma unroll
    for (int nt = 0; nt < 4; ++nt) acc[mt][nt] = zero8();

  const int rowBase = blockIdx.x * 128;
  const int colBase = blockIdx.y * 128;
  gemm_tile(ao, wp, C_DIM, rowBase, colBase,
            sA[0], sB[0], sA[1], sB[1], acc);

  const int tid = threadIdx.x, lane = tid & 31, wave = tid >> 5;
  const int wm = wave >> 1, wn = wave & 1;
  const int half = lane >> 4, l15 = lane & 15;
  const int mB = rowBase + wm * 32;
  const int nB = colBase + wn * 64;
#pragma unroll
  for (int mt = 0; mt < 2; ++mt)
#pragma unroll
    for (int r = 0; r < 8; ++r) {
      const int m = mB + mt * 16 + half * 8 + r;
#pragma unroll
      for (int nt = 0; nt < 4; ++nt) {
        const int col = nB + nt * 16 + l15;
        out[(size_t)m * C_DIM + col] = acc[mt][nt][r] + bias[col];
      }
    }
}

// ---------------------------------------------------------------------------
// Launch
// ---------------------------------------------------------------------------
extern "C" void kernel_launch(void* const* d_in, const int* in_sizes, int n_in,
                              void* d_out, int out_size, void* d_ws, size_t ws_size,
                              hipStream_t stream) {
  const float* x      = (const float*)d_in[0];
  const float* w_qkv  = (const float*)d_in[1];
  const float* w_proj = (const float*)d_in[2];
  const float* b_proj = (const float*)d_in[3];
  // d_in[4]/d_in[5] = H, W (fixed at 56; ignored)

  char* ws = (char*)d_ws;
  __bf16* wqkv_bf  = (__bf16*)(ws);                 // 1536*512  bf16 = 1,572,864 B
  __bf16* wproj_bf = (__bf16*)(ws + 1572864);       //  512*512  bf16 =   524,288 B
  __bf16* xb       = (__bf16*)(ws + 2097152);       // 32*3136*512 bf16 = 102,760,448 B
  __bf16* qws      = (__bf16*)(ws + 104857600);     // 32*64*8*64*64 bf16 = 134,217,728 B
  __bf16* kws      = (__bf16*)(ws + 239075328);
  __bf16* vws      = (__bf16*)(ws + 373293056);
  __bf16* aows     = xb;                            // xb is dead after QKV GEMM

  cvt_weights_kernel<<<4096, 256, 0, stream>>>(w_qkv, w_proj, wqkv_bf, wproj_bf);
  cvt_x_kernel<<<MTOK * C_DIM / 2048, 256, 0, stream>>>(x, xb);
  qkv_rope_kernel<<<dim3(MTOK / 128, NT3 / 128), 256, 0, stream>>>(
      xb, wqkv_bf, qws, kws, vws);
  attn_kernel<<<32 * 64 * 8, 128, 0, stream>>>(qws, kws, vws, aows);
  proj_kernel<<<dim3(MTOK / 128, C_DIM / 128), 256, 0, stream>>>(
      aows, wproj_bf, b_proj, (float*)d_out);
}
